// MDTA_5531917877953
// MI455X (gfx1250) — matrix operand order repaired and checked by reading, not compiled, not run on hardware
//
#include <hip/hip_runtime.h>
#include <hip/hip_bf16.h>
#include <math.h>

// ---------------- problem constants ----------------
#define BDIM   4
#define CDIM   384
#define NHEADS 8
#define CHEAD  48            // CDIM / NHEADS
#define HH     128
#define WW     128
#define HWN    (HH*WW)       // 16384
#define OC3    (3*CDIM)      // 1152

// ---------------- WMMA types ----------------
typedef __attribute__((ext_vector_type(16))) __bf16 v16bf;
typedef __attribute__((ext_vector_type(8)))  float  v8f;
typedef __attribute__((ext_vector_type(4)))  unsigned int u32x4;

union ABu { v16bf v; u32x4 u[2]; };

static __device__ __forceinline__ v8f wmma_bf16(const ABu& a, const ABu& b, v8f c) {
  // D = A(16x32 bf16) x B(32x16 bf16) + C(16x16 f32)
  return __builtin_amdgcn_wmma_f32_16x16x32_bf16(false, a.v, false, b.v,
                                                 (short)0, c, false, false);
}

// Per-lane fragment addressing as loaded here (wave32; A and B use the same k order):
//  A (16x32, MxK): lane = M | (hi<<4); elems j -> K = 16*hi + j   (contiguous 16)
//  B (32x16, KxN): lane = N | (hi<<4); elems j -> K = 16*hi + j   (contiguous 16)
//  C/D (16x16 f32): lane = N | (hi<<4); vgpr r -> M = r + 8*hi

// async global -> LDS copy of one 16B chunk per lane (ASYNCcnt-tracked)
static __device__ __forceinline__ void async_ld16(unsigned lds_off, const void* gaddr) {
  asm volatile("global_load_async_to_lds_b128 %0, %1, off"
               :: "v"(lds_off), "v"(gaddr) : "memory");
}
static __device__ __forceinline__ void wait_async0() {
  asm volatile("s_wait_asynccnt 0x0" ::: "memory");
}

// ---------------- elementwise f32 -> bf16 (weights) ----------------
__global__ void cvt_f32_bf16(const float* __restrict__ src, __bf16* __restrict__ dst, int n) {
  int i = blockIdx.x * 256 + threadIdx.x;
  if (i < n) dst[i] = (__bf16)src[i];
}

// ---------------- LDS-tiled transpose + convert:  [rows][cols] -> [cols][rows] bf16 ----
template <typename T>
__global__ void transpose_to_bf16(const T* __restrict__ src, __bf16* __restrict__ dst,
                                  size_t srcBatch, size_t dstBatch, int rows, int cols) {
  __shared__ T tile[32][33];
  const int b  = blockIdx.z;
  const int n0 = blockIdx.x * 32;
  const int r0 = blockIdx.y * 32;
  const T* s = src + (size_t)b * srcBatch;
  __bf16*  d = dst + (size_t)b * dstBatch;
#pragma unroll
  for (int i = 0; i < 4; ++i) {
    int r = r0 + threadIdx.y + i * 8;
    tile[threadIdx.y + i * 8][threadIdx.x] = s[(size_t)r * cols + n0 + threadIdx.x];
  }
  __syncthreads();
#pragma unroll
  for (int i = 0; i < 4; ++i) {
    int n = n0 + threadIdx.y + i * 8;
    d[(size_t)n * rows + r0 + threadIdx.x] = (__bf16)tile[threadIdx.x][threadIdx.y + i * 8];
  }
}

// ---------------- GEMM: Out[b][m][n] = sum_k W[m][k] * Xt[b][n][k] ----------------
// W: [M][K] bf16 row-major (K % 32 == 0), Xt: [B][N][K] bf16.
// Block: 256 thr = 8 waves -> 128(M) x 64(N); wave -> 16(M) x 64(N).
// B-tile (64 x 32 bf16 = 4KB) staged once per block per K-step via async
// global->LDS DMA, double-buffered; fragments read back with ds_load_b128.
// LDS row stride = 40 halfs (80B): rows 0..15 hit disjoint 4-bank groups.
#define BROW 40
template <typename OutT>
__global__ void gemm_lds(const __bf16* __restrict__ W, const __bf16* __restrict__ Xt,
                         OutT* __restrict__ Out, int M, int K, int N) {
  __shared__ __align__(16) __bf16 btile[2][64 * BROW];

  const int wave = threadIdx.x >> 5, lane = threadIdx.x & 31;
  const int hi = lane >> 4, l15 = lane & 15;
  const int b  = blockIdx.z;
  const int n0 = blockIdx.x * 64;
  const int m0 = blockIdx.y * 128 + wave * 16;

  const __bf16* wrow = W + (size_t)(m0 + l15) * K;
  // async-copy assignment: thread -> (row, 16B chunk) of the 64x32 tile
  const int crow = threadIdx.x >> 2, cchk = threadIdx.x & 3;
  const __bf16* gsrc0 = Xt + ((size_t)b * N + n0 + crow) * K + cchk * 8;
  const unsigned loff = (unsigned)(size_t)&btile[0][crow * BROW + cchk * 8];
  const unsigned lstride = (unsigned)(64 * BROW * 2);   // bytes between buffers

  // prologue: fetch K-step 0 into buffer 0
  async_ld16(loff, (const void*)gsrc0);

  v8f acc[4] = {};
  int cur = 0;
  for (int k = 0; k < K; k += 32) {
    wait_async0();          // this wave's chunks for buf[cur] have landed
    __syncthreads();        // whole tile in LDS; everyone done reading buf[cur^1]
    if (k + 32 < K)         // overlap: DMA next tile while computing this one
      async_ld16(loff + (cur ^ 1) * lstride, (const void*)(gsrc0 + k + 32));

    ABu a;
    a.u[0] = *(const u32x4*)(wrow + k + 16 * hi);
    a.u[1] = *(const u32x4*)(wrow + k + 16 * hi + 8);
#pragma unroll
    for (int nt = 0; nt < 4; ++nt) {
      const __bf16* lp = &btile[cur][(nt * 16 + l15) * BROW + 16 * hi];
      ABu bb;
      bb.u[0] = *(const u32x4*)(lp);
      bb.u[1] = *(const u32x4*)(lp + 8);
      acc[nt] = wmma_bf16(a, bb, acc[nt]);
    }
    cur ^= 1;
  }
#pragma unroll
  for (int nt = 0; nt < 4; ++nt)
#pragma unroll
    for (int r = 0; r < 8; ++r)
      Out[((size_t)b * M + m0 + r + 8 * hi) * N + n0 + nt * 16 + l15] = (OutT)acc[nt][r];
}

// ---------------- depthwise 3x3, pad=1, groups=1152 ----------------
__global__ void dwconv3x3(const __bf16* __restrict__ in, const float* __restrict__ wdw,
                          __bf16* __restrict__ out) {
  const int ch = blockIdx.y, b = blockIdx.z;
  const int pix = blockIdx.x * 256 + threadIdx.x;   // 0..16383
  const int x = pix & (WW - 1), y = pix >> 7;
  const __bf16* base = in + ((size_t)b * OC3 + ch) * HWN;
  float wv[9];
#pragma unroll
  for (int i = 0; i < 9; ++i) wv[i] = wdw[ch * 9 + i];
  float s = 0.f;
#pragma unroll
  for (int dy = -1; dy <= 1; ++dy)
#pragma unroll
    for (int dx = -1; dx <= 1; ++dx) {
      int yy = y + dy, xx = x + dx;
      if (yy >= 0 && yy < HH && xx >= 0 && xx < WW)
        s += wv[(dy + 1) * 3 + (dx + 1)] * (float)base[yy * WW + xx];
    }
  out[((size_t)b * OC3 + ch) * HWN + pix] = (__bf16)s;
}

// ---------------- per-channel 1/max(||row||,eps) for q,k (channels 0..767) ----------
__global__ void rowsumsq(const __bf16* __restrict__ qkv, float* __restrict__ scales) {
  const int ch = blockIdx.x, b = blockIdx.y;
  const __bf16* p = qkv + ((size_t)b * OC3 + ch) * HWN;
  float s = 0.f;
  for (int i = threadIdx.x * 8; i < HWN; i += 256 * 8) {
    union { u32x4 u; __bf16 h[8]; } c;
    c.u = *(const u32x4*)(p + i);
#pragma unroll
    for (int j = 0; j < 8; ++j) { float v = (float)c.h[j]; s += v * v; }
  }
  __shared__ float red[256];
  red[threadIdx.x] = s; __syncthreads();
  for (int off = 128; off > 0; off >>= 1) {
    if (threadIdx.x < off) red[threadIdx.x] += red[threadIdx.x + off];
    __syncthreads();
  }
  if (threadIdx.x == 0)
    scales[(size_t)b * 768 + ch] = 1.f / fmaxf(sqrtf(red[0]), 1e-12f);
}

// ---------------- attention: S = q_n k_n^T * T; P = softmax(S); outT = (P v)^T ------
// one block per (head, batch); 8 waves split K=16384 (phase 1) / N=16384 (phase 2)
__global__ void attn_kernel(const __bf16* __restrict__ qkv,   // [B][1152][HWN] post-dwconv
                            const __bf16* __restrict__ vt,    // [B][HWN][384]
                            const float* __restrict__ scales, // [B][768]
                            const float* __restrict__ temp,   // [8]
                            __bf16* __restrict__ outT) {      // [B][HWN][384]
  const int h = blockIdx.x, b = blockIdx.y;
  const int tid = threadIdx.x, wave = tid >> 5, lane = tid & 31;
  const int hi = lane >> 4, l15 = lane & 15;

  __shared__ __align__(16) float  S[CHEAD][CHEAD];
  __shared__ __align__(16) __bf16 P[CHEAD][64];

  for (int i = tid; i < CHEAD * CHEAD; i += 256) ((float*)S)[i] = 0.f;
  __syncthreads();

  // ---- phase 1: S[m][d] += sum_n q[m][n] k[d][n] over this wave's K slice ----
  const size_t qbase = ((size_t)b * OC3 + h * CHEAD) * HWN;          // q chans
  const size_t kbase = ((size_t)b * OC3 + CDIM + h * CHEAD) * HWN;   // k chans
  v8f acc[3][3] = {};
  const int k0 = wave * 2048;
  for (int k = k0; k < k0 + 2048; k += 32) {
    ABu a[3], bf[3];
#pragma unroll
    for (int mt = 0; mt < 3; ++mt) {
      const __bf16* p = qkv + qbase + (size_t)(mt * 16 + l15) * HWN + k;
      a[mt].u[0] = *(const u32x4*)(p + 16 * hi);
      a[mt].u[1] = *(const u32x4*)(p + 16 * hi + 8);
    }
#pragma unroll
    for (int nt = 0; nt < 3; ++nt) {
      const __bf16* p = qkv + kbase + (size_t)(nt * 16 + l15) * HWN + k + 16 * hi;
      bf[nt].u[0] = *(const u32x4*)(p);
      bf[nt].u[1] = *(const u32x4*)(p + 8);
    }
#pragma unroll
    for (int mt = 0; mt < 3; ++mt)
#pragma unroll
      for (int nt = 0; nt < 3; ++nt)
        acc[mt][nt] = wmma_bf16(a[mt], bf[nt], acc[mt][nt]);
  }
  const float tmprt = temp[h];
#pragma unroll
  for (int mt = 0; mt < 3; ++mt)
#pragma unroll
    for (int nt = 0; nt < 3; ++nt)
#pragma unroll
      for (int r = 0; r < 8; ++r) {
        int m = mt * 16 + r + 8 * hi;
        int d = nt * 16 + l15;
        float qs = scales[(size_t)b * 768 + h * CHEAD + m];
        float ks = scales[(size_t)b * 768 + CDIM + h * CHEAD + d];
        atomicAdd(&S[m][d], acc[mt][nt][r] * qs * ks * tmprt);
      }
  __syncthreads();

  // ---- softmax rows of S -> bf16 P (padded to 64 cols with zeros) ----
  if (tid < CHEAD) {
    const int m = tid;
    float mx = -1e30f;
    for (int d = 0; d < CHEAD; ++d) mx = fmaxf(mx, S[m][d]);
    float e[CHEAD], sum = 0.f;
    for (int d = 0; d < CHEAD; ++d) { e[d] = __expf(S[m][d] - mx); sum += e[d]; }
    float inv = 1.f / sum;
    for (int d = 0; d < CHEAD; ++d) P[m][d] = (__bf16)(e[d] * inv);
    for (int d = CHEAD; d < 64; ++d) P[m][d] = (__bf16)0.f;
  }
  __syncthreads();

  // ---- phase 2: outT[n][h*48 + c] = sum_d P[c][d] * v[d][n]  (K padded to 64) ----
  ABu pa[3][2];
#pragma unroll
  for (int mt = 0; mt < 3; ++mt)
#pragma unroll
    for (int kkI = 0; kkI < 2; ++kkI) {
      const __bf16* p = &P[mt * 16 + l15][kkI * 32];
      pa[mt][kkI].u[0] = *(const u32x4*)(p + 16 * hi);
      pa[mt][kkI].u[1] = *(const u32x4*)(p + 16 * hi + 8);
    }
  const int nb0 = wave * 2048;
  for (int n = nb0; n < nb0 + 2048; n += 16) {
    const __bf16* vp = vt + ((size_t)b * HWN + n + l15) * CDIM + h * CHEAD;
    ABu vb[2];
#pragma unroll
    for (int kkI = 0; kkI < 2; ++kkI) {
      const __bf16* p = vp + kkI * 32 + 16 * hi;     // cols >=48 hit pad; P there is 0
      vb[kkI].u[0] = *(const u32x4*)(p);
      vb[kkI].u[1] = *(const u32x4*)(p + 8);
    }
    v8f o[3] = {};
#pragma unroll
    for (int mt = 0; mt < 3; ++mt) {
      o[mt] = wmma_bf16(pa[mt][0], vb[0], o[mt]);
      o[mt] = wmma_bf16(pa[mt][1], vb[1], o[mt]);
    }
#pragma unroll
    for (int mt = 0; mt < 3; ++mt) {
      union { __bf16 hx[8]; u32x4 u; } pk;
#pragma unroll
      for (int r = 0; r < 8; ++r) pk.hx[r] = (__bf16)o[mt][r];
      *(u32x4*)(outT + ((size_t)b * HWN + n + l15) * CDIM + h * CHEAD + mt * 16 + 8 * hi) = pk.u;
    }
  }
}

// ---------------- host side ----------------
extern "C" void kernel_launch(void* const* d_in, const int* in_sizes, int n_in,
                              void* d_out, int out_size, void* d_ws, size_t ws_size,
                              hipStream_t stream) {
  (void)in_sizes; (void)n_in; (void)out_size; (void)ws_size;
  const float* x      = (const float*)d_in[0];   // [4][384][128][128]
  const float* w_qkv  = (const float*)d_in[1];   // [1152][384]
  const float* w_dw   = (const float*)d_in[2];   // [1152][1][3][3]
  const float* w_proj = (const float*)d_in[3];   // [384][384]
  const float* temp   = (const float*)d_in[4];   // [8]
  float* out = (float*)d_out;                    // [4][384][128][128]

  // workspace layout (bytes, all 256-aligned)
  char* ws = (char*)d_ws;
  size_t off = 0;
  __bf16* wqkv_b  = (__bf16*)(ws + off); off += (size_t)OC3 * CDIM * 2;            // 0.88 MB
  __bf16* wproj_b = (__bf16*)(ws + off); off += (size_t)CDIM * CDIM * 2;           // 0.29 MB
  __bf16* xt      = (__bf16*)(ws + off); off += (size_t)BDIM * HWN * CDIM * 2;     // 50 MB
  __bf16* qkv_a   = (__bf16*)(ws + off); off += (size_t)BDIM * OC3 * HWN * 2;      // 151 MB
  __bf16* qkv_b   = (__bf16*)(ws + off); off += (size_t)BDIM * OC3 * HWN * 2;      // 151 MB
  __bf16* vtr     = (__bf16*)(ws + off); off += (size_t)BDIM * HWN * CDIM * 2 + 256; // 50 MB (+pad)
  float*  scales  = (float*)(ws + off);  off += (size_t)BDIM * 768 * 4;            // 12 KB
  __bf16* aoutT   = (__bf16*)(ws + off); off += (size_t)BDIM * HWN * CDIM * 2;     // 50 MB

  // 1) weight conversions to bf16
  cvt_f32_bf16<<<(OC3 * CDIM + 255) / 256, 256, 0, stream>>>(w_qkv, wqkv_b, OC3 * CDIM);
  cvt_f32_bf16<<<(CDIM * CDIM + 255) / 256, 256, 0, stream>>>(w_proj, wproj_b, CDIM * CDIM);

  // 2) x [b][c][n] f32 -> xt [b][n][c] bf16
  transpose_to_bf16<float><<<dim3(HWN / 32, CDIM / 32, BDIM), dim3(32, 8), 0, stream>>>(
      x, xt, (size_t)CDIM * HWN, (size_t)HWN * CDIM, CDIM, HWN);

  // 3) qkv = W_qkv @ x   (bf16 WMMA, f32 acc, async-LDS staged B) -> qkv_a [b][1152][n]
  gemm_lds<__bf16><<<dim3(HWN / 64, OC3 / 128, BDIM), 256, 0, stream>>>(
      wqkv_b, xt, qkv_a, OC3, CDIM, HWN);

  // 4) depthwise 3x3 -> qkv_b
  dwconv3x3<<<dim3(HWN / 256, OC3, BDIM), 256, 0, stream>>>(qkv_a, w_dw, qkv_b);

  // 5) q,k channel norms -> scales
  rowsumsq<<<dim3(768, BDIM), 256, 0, stream>>>(qkv_b, scales);

  // 6) v [b][384][n] -> vtr [b][n][384]
  transpose_to_bf16<__bf16><<<dim3(HWN / 32, CDIM / 32, BDIM), dim3(32, 8), 0, stream>>>(
      qkv_b + (size_t)2 * CDIM * HWN, vtr, (size_t)OC3 * HWN, (size_t)HWN * CDIM, CDIM, HWN);

  // 7) attention (per b,head): softmax(q_n k_n^T * T) @ v -> aoutT [b][n][c]
  attn_kernel<<<dim3(NHEADS, BDIM), 256, 0, stream>>>(qkv_b, vtr, scales, temp, aoutT);

  // 8) final projection: out = W_proj @ attn_out  (f32 output)
  gemm_lds<float><<<dim3(HWN / 64, CDIM / 128, BDIM), 256, 0, stream>>>(
      wproj_b, aoutT, out, CDIM, CDIM, HWN);
}
